// OneDirectionalTreeLSTM_67190468378863
// MI455X (gfx1250) — matrix operand order repaired and checked by reading, not compiled, not run on hardware
//
#include <hip/hip_runtime.h>
#include <hip/hip_bf16.h>

// ---------------------------------------------------------------------------
// Tree-LSTM (root->leaves), level-batched, bf16 WMMA GEMMs on gfx1250.
//   N = 50000 nodes, F = H = 512, BRANCH = 8.
//   Per level: iofu = [feats|ph] @ [W_iofux|W_iofuh].T + (b_iofux+b_iofuh)
//              px   = feats @ W_px.T + b_px
//              gates -> c, h, h_final ; h_final is both output and child input.
// ---------------------------------------------------------------------------

typedef __attribute__((ext_vector_type(16))) __bf16 v16bf;
typedef __attribute__((ext_vector_type(8)))  float  v8f;

#define HDIM   512
#define KDIM   1024           // 512 feats + 512 parent-h
#define GCOLS  2560           // 5 * H gate columns

__device__ __forceinline__ float sigmoidf_(float x) {
    return 1.0f / (1.0f + __expf(-x));
}

// ---------------- prep kernels -------------------------------------------

__global__ void k_f32_to_bf16(const float* __restrict__ src,
                              __bf16* __restrict__ dst, int n) {
    int i = blockIdx.x * blockDim.x + threadIdx.x;
    if (i < n) dst[i] = (__bf16)src[i];
}

// Wcat[c][k] = k<512 ? W_iofux[c][k] : W_iofuh[c][k-512]   (c < 2560, k < 1024)
__global__ void k_build_wcat(const float* __restrict__ Wx,
                             const float* __restrict__ Wh,
                             __bf16* __restrict__ out, int n) {
    int i = blockIdx.x * blockDim.x + threadIdx.x;
    if (i >= n) return;
    int c = i >> 10;
    int k = i & 1023;
    float v = (k < 512) ? Wx[c * 512 + k] : Wh[c * 512 + (k - 512)];
    out[i] = (__bf16)v;
}

// zero sentinel "parent of root" rows: h_bf[N], c_ws[N]
__global__ void k_zero_root_parent(__bf16* __restrict__ h_bf,
                                   float* __restrict__ c_ws, int N) {
    int j = blockIdx.x * blockDim.x + threadIdx.x;
    if (j < HDIM) {
        h_bf[(size_t)N * HDIM + j] = (__bf16)0.0f;
        c_ws[(size_t)N * HDIM + j] = 0.0f;
    }
}

// ---------------- per-level fused GEMM + gates ---------------------------
// block: 256 threads = 8 wave32.  Tile: 16 rows x 256 hidden units.
// grid = ( ceil((e-s)/16), 2 )  -> grid.y covers hidden units 0..511.
// Each wave: 16 rows x 32 hidden units; six K-loop passes (i,o,f,u,r,px),
// two v_wmma_f32_16x16x32_bf16 per K-step.  #pragma unroll 4 bounds the
// software-pipeline depth so the wave stays under 256 VGPRs.

__global__ __launch_bounds__(256)
void k_tree_level(const __bf16* __restrict__ feats_bf,
                  const __bf16* __restrict__ Wcat,   // [2560][1024] bf16
                  const __bf16* __restrict__ Wpx,    // [512][512]  bf16
                  const float*  __restrict__ b_iofux,
                  const float*  __restrict__ b_iofuh,
                  const float*  __restrict__ b_px,
                  float*  __restrict__ c_ws,         // [(N+1)][512] f32 (rw)
                  __bf16* __restrict__ h_bf,         // [(N+1)][512] bf16 (rw)
                  float*  __restrict__ h_out,        // [N][512] f32 (d_out)
                  int s, int e, int N) {
    __shared__ __bf16 As[16][KDIM];                  // 32 KB A tile

    const int rowBase = s + blockIdx.x * 16;

    // ---- stage A tile: row m = [feats[node] | h[parent(node)]] ----
    {
        int t  = threadIdx.x;
        int r  = t >> 4;                 // 0..15 row within tile
        int cb = (t & 15) * 64;          // 64 bf16 per thread, one source each
        int node = rowBase + r;
        if (node >= e) node = e - 1;     // padded rows: clamp (discarded later)
        const __bf16* src;
        if (cb < 512) {
            src = feats_bf + (size_t)node * HDIM + cb;
        } else {
            long par = (node == 0) ? (long)N : (long)(node - 1) >> 3;
            src = h_bf + (size_t)par * HDIM + (cb - 512);
        }
        uint4*       dq = (uint4*)&As[r][cb];
        const uint4* sq = (const uint4*)src;
#pragma unroll
        for (int q = 0; q < 8; ++q) dq[q] = sq[q];
    }
    __syncthreads();

    const int lane  = threadIdx.x & 31;
    const int wave  = threadIdx.x >> 5;
    const int hi    = lane >> 4;         // half-wave select
    const int n15   = lane & 15;         // row (A) / column (B,C,D) index
    const int jbase = blockIdx.y * 256 + wave * 32;   // hidden-unit tile base

    // gacc[g][t][v]: g = i,o,f,u,r,px ; t = 16-col sub-tile ; v = C/D VGPR
    float gacc[6][2][8];

#pragma unroll
    for (int g = 0; g < 6; ++g) {
        v8f acc0 = {};
        v8f acc1 = {};
        const __bf16* Brow0;
        const __bf16* Brow1;
        int Klen;
        if (g < 5) {                     // gate columns from Wcat, K = 1024
            Klen  = KDIM;
            Brow0 = Wcat + (size_t)(g * HDIM + jbase + n15)      * KDIM;
            Brow1 = Wcat + (size_t)(g * HDIM + jbase + 16 + n15) * KDIM;
        } else {                         // px columns from Wpx, K = 512
            Klen  = 512;
            Brow0 = Wpx + (size_t)(jbase + n15)      * 512;
            Brow1 = Wpx + (size_t)(jbase + 16 + n15) * 512;
        }
#pragma unroll 4
        for (int kb = 0; kb < Klen; kb += 32) {
            // A fragment (16x32 bf16): lane m=n15 needs K = kb+hi*16+[0..15]
            // -> two 16B LDS chunks.
            union { v16bf v; uint4 q[2]; } a, b0, b1;
            a.q[0] = *(const uint4*)&As[n15][kb + hi * 16];
            a.q[1] = *(const uint4*)&As[n15][kb + hi * 16 + 8];
            // B fragment (32x16 bf16): lane col = n15(+16), K = kb+hi*16+[0..15]
            // -> one 32B contiguous weight-row chunk.
            const uint4* bq0 = (const uint4*)(Brow0 + kb + hi * 16);
            b0.q[0] = bq0[0]; b0.q[1] = bq0[1];
            const uint4* bq1 = (const uint4*)(Brow1 + kb + hi * 16);
            b1.q[0] = bq1[0]; b1.q[1] = bq1[1];

            acc0 = __builtin_amdgcn_wmma_f32_16x16x32_bf16(
                       false, a.v, false, b0.v, (short)0, acc0, false, false);
            acc1 = __builtin_amdgcn_wmma_f32_16x16x32_bf16(
                       false, a.v, false, b1.v, (short)0, acc1, false, false);
        }
#pragma unroll
        for (int v = 0; v < 8; ++v) {
            gacc[g][0][v] = acc0[v];
            gacc[g][1][v] = acc1[v];
        }
    }

    // ---- epilogue: gates, parent-c gather, stores ----
#pragma unroll
    for (int t = 0; t < 2; ++t) {
        const int j = jbase + t * 16 + n15;            // hidden unit (C/D col)
        const float bi = b_iofux[j]        + b_iofuh[j];
        const float bo = b_iofux[512 + j]  + b_iofuh[512 + j];
        const float bf = b_iofux[1024 + j] + b_iofuh[1024 + j];
        const float bu = b_iofux[1536 + j] + b_iofuh[1536 + j];
        const float br = b_iofux[2048 + j] + b_iofuh[2048 + j];
        const float bp = b_px[j];
#pragma unroll
        for (int v = 0; v < 8; ++v) {
            const int m    = v + hi * 8;               // C/D row within tile
            const int node = rowBase + m;
            const bool ok  = (node < e);
            const int  nn  = ok ? node : (e - 1);
            const long par = (nn == 0) ? (long)N : (long)(nn - 1) >> 3;
            const float pc = c_ws[par * HDIM + j];

            const float xi = sigmoidf_(gacc[0][t][v] + bi);
            const float xo = sigmoidf_(gacc[1][t][v] + bo);
            const float xf = sigmoidf_(gacc[2][t][v] + bf);
            const float xu = tanhf    (gacc[3][t][v] + bu);
            const float xr = sigmoidf_(gacc[4][t][v] + br);
            const float px = gacc[5][t][v] + bp;

            const float c  = xi * xu + xf * pc;
            const float h  = xo * tanhf(c);
            const float hf = xr * h + (1.0f - xr) * px;

            if (ok) {
                const size_t off = (size_t)node * HDIM + j;
                h_out[off] = hf;
                c_ws[off]  = c;
                h_bf[off]  = (__bf16)hf;
            }
        }
    }
}

// ---------------------------------------------------------------------------

extern "C" void kernel_launch(void* const* d_in, const int* in_sizes, int n_in,
                              void* d_out, int out_size, void* d_ws, size_t ws_size,
                              hipStream_t stream) {
    const float* feats   = (const float*)d_in[0];
    const float* W_iofux = (const float*)d_in[1];
    const float* b_iofux = (const float*)d_in[2];
    const float* W_iofuh = (const float*)d_in[3];
    const float* b_iofuh = (const float*)d_in[4];
    const float* W_px    = (const float*)d_in[5];
    const float* b_px    = (const float*)d_in[6];
    float*       h_out   = (float*)d_out;

    const int N = in_sizes[0] / HDIM;    // 50000

    // ---- carve workspace ----
    char* p = (char*)d_ws;
    auto take = [&](size_t bytes) {
        char* q = p;
        p += (bytes + 255) & ~(size_t)255;
        return q;
    };
    __bf16* feats_bf = (__bf16*)take((size_t)N * HDIM * sizeof(__bf16));
    __bf16* h_bf     = (__bf16*)take((size_t)(N + 1) * HDIM * sizeof(__bf16));
    float*  c_ws     = (float*)take((size_t)(N + 1) * HDIM * sizeof(float));
    __bf16* Wcat     = (__bf16*)take((size_t)GCOLS * KDIM * sizeof(__bf16));
    __bf16* Wpx      = (__bf16*)take((size_t)HDIM * HDIM * sizeof(__bf16));
    (void)ws_size;

    // ---- prep ----
    {
        int n = N * HDIM;
        k_f32_to_bf16<<<(n + 255) / 256, 256, 0, stream>>>(feats, feats_bf, n);
    }
    {
        int n = GCOLS * KDIM;
        k_build_wcat<<<(n + 255) / 256, 256, 0, stream>>>(W_iofux, W_iofuh, Wcat, n);
    }
    {
        int n = HDIM * HDIM;
        k_f32_to_bf16<<<(n + 255) / 256, 256, 0, stream>>>(W_px, Wpx, n);
    }
    k_zero_root_parent<<<2, 256, 0, stream>>>(h_bf, c_ws, N);

    // ---- level bounds of complete 8-ary tree ----
    int bounds[16];
    int nb = 0;
    bounds[nb++] = 0;
    long cnt = 1;
    while (bounds[nb - 1] < N) {
        long nxt = (long)bounds[nb - 1] + cnt;
        bounds[nb++] = (int)(nxt < N ? nxt : N);
        cnt *= 8;
    }

    // ---- levels, serialized by stream order ----
    for (int l = 0; l < nb - 1; ++l) {
        const int s = bounds[l];
        const int e = bounds[l + 1];
        const int rows = e - s;
        dim3 grid((rows + 15) / 16, 2);
        k_tree_level<<<grid, 256, 0, stream>>>(
            feats_bf, Wcat, Wpx, b_iofux, b_iofuh, b_px,
            c_ws, h_bf, h_out, s, e, N);
    }
}
